// NeuralMemory_89781996355758
// MI455X (gfx1250) — matrix operand order repaired and checked by reading, not compiled, not run on hardware
//
#include <hip/hip_runtime.h>

#define BDIM 4
#define SDIM 512
#define DDIM 512
#define IDIM 1024
#define CDIM 64
#define NCH 8
#define ROWS_ALL (BDIM*SDIM)   // 2048
#define ROWS_CH  (BDIM*CDIM)   // 256
#define EPSV 1e-6f

typedef __attribute__((ext_vector_type(16))) __bf16 v16bf;
typedef __attribute__((ext_vector_type(8)))  float v8f;
typedef unsigned short u16_t;

union Frag16 { v16bf v; uint4 q[2]; };

__device__ __forceinline__ u16_t f2bf(float x) {
  unsigned int u = __float_as_uint(x);
  u += 0x7FFFu + ((u >> 16) & 1u);          // round-to-nearest-even
  return (u16_t)(u >> 16);
}
__device__ __forceinline__ float sigm(float x) { return 1.f / (1.f + __expf(-x)); }
__device__ __forceinline__ float siluf(float x) { return x * sigm(x); }

// ---------------------------------------------------------------------------
// bf16 NN GEMM: C[M,N] = A[M,K] * B[K,N], A/B row-major bf16, C row-major f32.
// Block = 256 threads = 8 waves arranged 4(M) x 2(N); wave tile 32x64
// (2x4 WMMA 16x16x32 -> 8 WMMAs per 4 b128 A loads + 64 scalar B loads); block tile 128x128.
// Requires M%128==0, N%128==0, K%32==0 (always true here).
// A frag (16-bit 16x32): lane l -> row m0+(l&15); K elems = kb..kb+7,
// kb+16..kb+23 with kb=8*(l>>4)  -> two 16B contiguous loads.
// B frag (16-bit 32x16): lane l -> col N = n0+(l&15); K elems = kb..kb+7,
// kb+16..kb+23 with kb=8*(l>>4) -> strided scalar loads.
// C/D layout: lanes 0-15: VGPR j -> (M=j, N=lane); lanes 16-31: (M=8+j).
// ---------------------------------------------------------------------------
__global__ void __launch_bounds__(256) gemm_bf16_nn(
    const u16_t* __restrict__ A, int lda,
    const u16_t* __restrict__ B, int ldb,
    float* __restrict__ C, int ldc, int K) {
  const int lane = threadIdx.x & 31;
  const int wave = threadIdx.x >> 5;
  const int m0 = blockIdx.x * 128 + (wave >> 1) * 32;
  const int n0 = blockIdx.y * 128 + (wave &  1) * 64;

  const u16_t* pa0 = A + (size_t)(m0 + (lane & 15)) * lda + ((lane >> 4) << 3);
  const u16_t* pa1 = pa0 + (size_t)16 * lda;
  const u16_t* pb0 = B + (size_t)((lane >> 4) << 3) * ldb + n0 + (lane & 15);

  v8f acc0[4] = {v8f{}, v8f{}, v8f{}, v8f{}};
  v8f acc1[4] = {v8f{}, v8f{}, v8f{}, v8f{}};

  for (int k0 = 0; k0 < K; k0 += 32) {
    Frag16 a0, a1, b[4];
    a0.q[0] = *(const uint4*)(pa0 + k0);
    a0.q[1] = *(const uint4*)(pa0 + k0 + 16);
    a1.q[0] = *(const uint4*)(pa1 + k0);
    a1.q[1] = *(const uint4*)(pa1 + k0 + 16);
    const u16_t* pb = pb0 + (size_t)k0 * ldb;
#pragma unroll
    for (int j = 0; j < 4; ++j) {
#pragma unroll
      for (int e = 0; e < 8; ++e) {
        b[j].v[e]     = __builtin_bit_cast(__bf16, pb[(size_t)e * ldb + 16 * j]);
        b[j].v[8 + e] = __builtin_bit_cast(__bf16, pb[(size_t)(16 + e) * ldb + 16 * j]);
      }
    }
    if (k0 + 32 < K) {                      // lowers to global_prefetch_b8
      __builtin_prefetch(pa0 + k0 + 32, 0, 3);
      __builtin_prefetch(pb0 + (size_t)(k0 + 32) * ldb, 0, 3);
    }
#pragma unroll
    for (int j = 0; j < 4; ++j) {
      acc0[j] = __builtin_amdgcn_wmma_f32_16x16x32_bf16(false, a0.v, false, b[j].v, (short)0, acc0[j], false, false);
      acc1[j] = __builtin_amdgcn_wmma_f32_16x16x32_bf16(false, a1.v, false, b[j].v, (short)0, acc1[j], false, false);
    }
  }
  const int cn = lane & 15;
  const int cm = (lane >> 4) << 3;
  float* c0 = C + (size_t)(m0 + cm) * ldc + n0 + cn;
  float* c1 = C + (size_t)(m0 + 16 + cm) * ldc + n0 + cn;
#pragma unroll
  for (int j = 0; j < 4; ++j) {
#pragma unroll
    for (int e = 0; e < 8; ++e) {
      c0[(size_t)e * ldc + 16 * j] = acc0[j][e];
      c1[(size_t)e * ldc + 16 * j] = acc1[j][e];
    }
  }
}

// ---------------------------------------------------------------------------
// One-time prep: bf16 transposed projection weights, param init (M=input,
// S=0), bf16 copies (w0T [D,I], w1 [D,I], w1T [I,D]).
// ---------------------------------------------------------------------------
__global__ void prep_kernel(const float* __restrict__ wq, const float* __restrict__ wk,
                            const float* __restrict__ wv,
                            const float* __restrict__ m_w0, const float* __restrict__ m_w1,
                            const float* __restrict__ m_ln,
                            u16_t* wqT, u16_t* wkT, u16_t* wvT,
                            float* Mw0, float* Sw0, u16_t* w0T,
                            float* Mw1, float* Sw1, u16_t* w1bf, u16_t* w1T,
                            float* Mln, float* Sln) {
  const size_t DD = (size_t)DDIM * DDIM, ID = (size_t)IDIM * DDIM;
  const size_t total = 3 * DD + 2 * ID + DDIM;
  for (size_t idx = (size_t)blockIdx.x * blockDim.x + threadIdx.x; idx < total;
       idx += (size_t)gridDim.x * blockDim.x) {
    if (idx < 3 * DD) {
      size_t which = idx / DD, j = idx % DD;
      size_t r = j / DDIM, ccol = j % DDIM;          // output [r][c] = W[c][r]
      const float* W = (which == 0) ? wq : (which == 1) ? wk : wv;
      u16_t* T = (which == 0) ? wqT : (which == 1) ? wkT : wvT;
      T[j] = f2bf(W[ccol * DDIM + r]);
    } else if (idx < 3 * DD + ID) {
      size_t j = idx - 3 * DD;                        // w0 [I,D]
      size_t i = j / DDIM, d = j % DDIM;
      float v = m_w0[j];
      Mw0[j] = v; Sw0[j] = 0.f;
      w0T[d * IDIM + i] = f2bf(v);
    } else if (idx < 3 * DD + 2 * ID) {
      size_t j = idx - 3 * DD - ID;                   // w1 [D,I]
      size_t d = j / IDIM, i = j % IDIM;
      float v = m_w1[j];
      Mw1[j] = v; Sw1[j] = 0.f;
      w1bf[j] = f2bf(v);
      w1T[i * DDIM + d] = f2bf(v);
    } else {
      size_t d = idx - 3 * DD - 2 * ID;
      Mln[d] = m_ln[d]; Sln[d] = 0.f;
    }
  }
}

// x [B,S,D] f32 -> xbf in chunk-major row order r = c*256 + b*64 + t
__global__ void pack_x_kernel(const float* __restrict__ x, u16_t* __restrict__ xbf) {
  size_t idx = (size_t)blockIdx.x * blockDim.x + threadIdx.x;
  if (idx >= (size_t)ROWS_ALL * DDIM) return;
  int r = (int)(idx / DDIM), d = (int)(idx % DDIM);
  int c = r >> 8, rem = r & 255, b = rem >> 6, t = rem & 63;
  int s = c * CDIM + t;
  xbf[idx] = f2bf(x[((size_t)b * SDIM + s) * DDIM + d]);
}

// ---------------------------------------------------------------------------
// Gates per chunk: al/th/et, then the collapsed scan coefficients:
//   gates[c*256 + 0..63]   = th_t (incl BASE_LR)
//   gates[c*256 + 64..127] = a_n   (weight of grad_n in newM)
//   gates[c*256 +128..191] = e_n   (weight of grad_n in newS)
//   [192]=bc_last  [193]=sA  [194]=sE
// ---------------------------------------------------------------------------
__global__ void __launch_bounds__(256) gates_kernel(
    const float* __restrict__ x,
    const float* __restrict__ aw, const float* __restrict__ ab,
    const float* __restrict__ tw, const float* __restrict__ tb,
    const float* __restrict__ ewv, const float* __restrict__ eb,
    float* __restrict__ gates) {
  const int c = blockIdx.x, tid = threadIdx.x;
  const int t = tid >> 2, b = tid & 3;
  const float* xr = x + ((size_t)b * SDIM + c * CDIM + t) * DDIM;
  float da = 0.f, dt = 0.f, de = 0.f;
  for (int d = 0; d < DDIM; ++d) {
    float xv = xr[d];
    da += xv * aw[d]; dt += xv * tw[d]; de += xv * ewv[d];
  }
  __shared__ float sa[256], st[256], se[256];
  __shared__ float al[64], th[64], et[64];
  sa[tid] = da; st[tid] = dt; se[tid] = de;
  __syncthreads();
  if (b == 0) {
    float A_ = sa[tid] + sa[tid+1] + sa[tid+2] + sa[tid+3] + 4.f * ab[0];
    float T_ = st[tid] + st[tid+1] + st[tid+2] + st[tid+3] + 4.f * tb[0];
    float E_ = se[tid] + se[tid+1] + se[tid+2] + se[tid+3] + 4.f * eb[0];
    al[t] = sigm(A_);
    th[t] = sigm(T_) * 0.01f;   // BASE_LR
    et[t] = sigm(E_);
  }
  __syncthreads();
  if (tid == 0) {
    float* g = gates + (size_t)c * 256;
    float eArr[64], wb[64], aArr[64];
    float prod = 1.f;
    for (int n = 63; n >= 0; --n) { eArr[n] = prod; prod *= et[n]; }
    float sE = prod;                                  // prod all eta
    float bprod = 1.f;
    for (int m = 63; m >= 0; --m) { wb[m] = bprod; bprod *= (1.f - al[m]); }
    float bc = bprod;                                 // prod all beta
    aArr[63] = wb[63];
    for (int n = 62; n >= 0; --n) aArr[n] = wb[n] + et[n+1] * aArr[n+1];
    float ce = 1.f, sA = 0.f;
    for (int m = 0; m < 64; ++m) { ce *= et[m]; sA += wb[m] * ce; }
    for (int n = 0; n < 64; ++n) { g[n] = th[n]; g[64+n] = aArr[n]; g[128+n] = eArr[n]; }
    g[192] = bc; g[193] = sA; g[194] = sE;
  }
}

// row-wise silu (+optional RMSNorm) over D=512; one block per row
__global__ void __launch_bounds__(256) silu_rms_kernel(
    const float* __restrict__ raw, const float* __restrict__ normw,
    float* __restrict__ outf, u16_t* __restrict__ outbf, int applyRms) {
  const int r = blockIdx.x, tid = threadIdx.x;
  const float* in = raw + (size_t)r * DDIM;
  float s0 = siluf(in[tid]), s1 = siluf(in[tid + 256]);
  float sc = 1.f;
  __shared__ float red[256];
  if (applyRms) {
    red[tid] = s0 * s0 + s1 * s1;
    __syncthreads();
    for (int off = 128; off > 0; off >>= 1) {
      if (tid < off) red[tid] += red[tid + off];
      __syncthreads();
    }
    sc = rsqrtf(red[0] / (float)DDIM + EPSV);
  }
  float o0 = s0 * sc, o1 = s1 * sc;
  if (applyRms) { o0 *= normw[tid]; o1 *= normw[tid + 256]; }
  if (outf)  { outf[(size_t)r * DDIM + tid] = o0; outf[(size_t)r * DDIM + tid + 256] = o1; }
  if (outbf) { outbf[(size_t)r * DDIM + tid] = f2bf(o0); outbf[(size_t)r * DDIM + tid + 256] = f2bf(o1); }
}

__global__ void silu_elem_kernel(const float* __restrict__ in, u16_t* __restrict__ outbf, size_t n) {
  size_t idx = (size_t)blockIdx.x * blockDim.x + threadIdx.x;
  if (idx < n) outbf[idx] = f2bf(siluf(in[idx]));
}

// ---------------------------------------------------------------------------
// RMS backward per row: pred = k + rms(y,ln); g = dL/dz = -(2*th/D)*(v-pred);
// dy = r*(g.ln) - (r^3/D)*(g.ln.y sum)*y ; dln = g*r*y.
// Writes dy (bf16, row-major), a/e-weighted dy transposed (bf16 [D,256]),
// and dln f32 for the ln-grad reduction.
// ---------------------------------------------------------------------------
__global__ void __launch_bounds__(256) rowback_kernel(
    const float* __restrict__ yraw, const float* __restrict__ kf,
    const float* __restrict__ vf, const float* __restrict__ lnw,
    const float* __restrict__ gc,
    u16_t* __restrict__ dybf, u16_t* __restrict__ dyTa, u16_t* __restrict__ dyTe,
    float* __restrict__ dln) {
  const int r = blockIdx.x, tid = threadIdx.x;
  const int t = r & (CDIM - 1);
  const float th = gc[t], awt = gc[64 + t], ewt = gc[128 + t];
  const float* y = yraw + (size_t)r * DDIM;
  const float* kk = kf + (size_t)r * DDIM;
  const float* vv = vf + (size_t)r * DDIM;
  float y0 = y[tid], y1 = y[tid + 256];
  float l0 = lnw[tid], l1 = lnw[tid + 256];
  __shared__ float red[256];
  red[tid] = y0 * y0 + y1 * y1;
  __syncthreads();
  for (int off = 128; off > 0; off >>= 1) { if (tid < off) red[tid] += red[tid + off]; __syncthreads(); }
  float rinv = rsqrtf(red[0] / (float)DDIM + EPSV);
  __syncthreads();
  const float cgl = -2.f * th / (float)DDIM;
  float g0 = cgl * (vv[tid]       - (kk[tid]       + y0 * rinv * l0));
  float g1 = cgl * (vv[tid + 256] - (kk[tid + 256] + y1 * rinv * l1));
  red[tid] = g0 * l0 * y0 + g1 * l1 * y1;
  __syncthreads();
  for (int off = 128; off > 0; off >>= 1) { if (tid < off) red[tid] += red[tid + off]; __syncthreads(); }
  float coef = rinv * rinv * rinv * (1.f / (float)DDIM) * red[0];
  float dy0 = rinv * g0 * l0 - coef * y0;
  float dy1 = rinv * g1 * l1 - coef * y1;
  dybf[(size_t)r * DDIM + tid]       = f2bf(dy0);
  dybf[(size_t)r * DDIM + tid + 256] = f2bf(dy1);
  dyTa[(size_t)tid * ROWS_CH + r]         = f2bf(awt * dy0);
  dyTa[(size_t)(tid + 256) * ROWS_CH + r] = f2bf(awt * dy1);
  dyTe[(size_t)tid * ROWS_CH + r]         = f2bf(ewt * dy0);
  dyTe[(size_t)(tid + 256) * ROWS_CH + r] = f2bf(ewt * dy1);
  dln[(size_t)r * DDIM + tid]       = g0 * rinv * y0;
  dln[(size_t)r * DDIM + tid + 256] = g1 * rinv * y1;
}

// da = dh * silu'(Araw); write a/e-weighted transposed bf16 [I,256]
__global__ void da_kernel(const float* __restrict__ Araw, const float* __restrict__ dh,
                          const float* __restrict__ gc,
                          u16_t* __restrict__ daTa, u16_t* __restrict__ daTe) {
  size_t idx = (size_t)blockIdx.x * blockDim.x + threadIdx.x;
  if (idx >= (size_t)ROWS_CH * IDIM) return;
  int r = (int)(idx >> 10), i = (int)(idx & (IDIM - 1));
  int t = r & 63;
  float a = Araw[idx];
  float sg = sigm(a);
  float sp = sg * (1.f + a * (1.f - sg));
  float d = dh[idx] * sp;
  daTa[(size_t)i * ROWS_CH + r] = f2bf(gc[64 + t] * d);
  daTe[(size_t)i * ROWS_CH + r] = f2bf(gc[128 + t] * d);
}

__global__ void reduce_dln_kernel(const float* __restrict__ dln, const float* __restrict__ gc,
                                  float* __restrict__ ulna, float* __restrict__ ulne) {
  int d = blockIdx.x * blockDim.x + threadIdx.x;
  if (d >= DDIM) return;
  float sa = 0.f, se = 0.f;
  for (int r = 0; r < ROWS_CH; ++r) {
    float v = dln[(size_t)r * DDIM + d];
    int t = r & 63;
    sa += gc[64 + t] * v;
    se += gc[128 + t] * v;
  }
  ulna[d] = sa; ulne[d] = se;
}

// newS = sE*S - Ue ; newM = bc*M + sA*S - Ua ; refresh bf16 weight copies
__global__ void update_kernel(const float* __restrict__ gc,
                              float* Mw0, float* Sw0, const float* __restrict__ U0a,
                              const float* __restrict__ U0e, u16_t* w0T,
                              float* Mw1, float* Sw1, const float* __restrict__ U1a,
                              const float* __restrict__ U1e, u16_t* w1bf, u16_t* w1T,
                              float* Mln, float* Sln, const float* __restrict__ ulna,
                              const float* __restrict__ ulne) {
  const float bc = gc[192], sA = gc[193], sE = gc[194];
  const size_t ID = (size_t)IDIM * DDIM;
  const size_t total = 2 * ID + DDIM;
  for (size_t idx = (size_t)blockIdx.x * blockDim.x + threadIdx.x; idx < total;
       idx += (size_t)gridDim.x * blockDim.x) {
    if (idx < ID) {
      float M = Mw0[idx], S = Sw0[idx];
      float nM = bc * M + sA * S - U0a[idx];
      float nS = sE * S - U0e[idx];
      Mw0[idx] = nM; Sw0[idx] = nS;
      size_t i = idx / DDIM, d = idx % DDIM;
      w0T[d * IDIM + i] = f2bf(nM);
    } else if (idx < 2 * ID) {
      size_t j = idx - ID;
      float M = Mw1[j], S = Sw1[j];
      float nM = bc * M + sA * S - U1a[j];
      float nS = sE * S - U1e[j];
      Mw1[j] = nM; Sw1[j] = nS;
      size_t d = j / IDIM, i = j % IDIM;
      w1bf[j] = f2bf(nM);
      w1T[i * DDIM + d] = f2bf(nM);
    } else {
      size_t d = idx - 2 * ID;
      float M = Mln[d], S = Sln[d];
      float nM = bc * M + sA * S - ulna[d];
      float nS = sE * S - ulne[d];
      Mln[d] = nM; Sln[d] = nS;
    }
  }
}

// out = q + rms(y2, lnF), mapped back from chunk-major rows to [B,S,D]
__global__ void __launch_bounds__(256) final_kernel(
    const float* __restrict__ yraw, const float* __restrict__ qf,
    const float* __restrict__ Mln, float* __restrict__ out) {
  const int r = blockIdx.x, tid = threadIdx.x;
  const float* y = yraw + (size_t)r * DDIM;
  float y0 = y[tid], y1 = y[tid + 256];
  __shared__ float red[256];
  red[tid] = y0 * y0 + y1 * y1;
  __syncthreads();
  for (int off = 128; off > 0; off >>= 1) { if (tid < off) red[tid] += red[tid + off]; __syncthreads(); }
  float rinv = rsqrtf(red[0] / (float)DDIM + EPSV);
  int c = r >> 8, rem = r & 255, b = rem >> 6, t = rem & 63;
  int s = c * CDIM + t;
  size_t o = ((size_t)b * SDIM + s) * DDIM;
  out[o + tid]       = qf[(size_t)r * DDIM + tid]       + y0 * rinv * Mln[tid];
  out[o + tid + 256] = qf[(size_t)r * DDIM + tid + 256] + y1 * rinv * Mln[tid + 256];
}

// ---------------------------------------------------------------------------
extern "C" void kernel_launch(void* const* d_in, const int* in_sizes, int n_in,
                              void* d_out, int out_size, void* d_ws, size_t ws_size,
                              hipStream_t stream) {
  (void)in_sizes; (void)n_in; (void)out_size; (void)ws_size;
  const float* x    = (const float*)d_in[0];
  const float* wq   = (const float*)d_in[1];
  const float* wk   = (const float*)d_in[2];
  const float* wv   = (const float*)d_in[3];
  const float* qnw  = (const float*)d_in[4];
  const float* knw  = (const float*)d_in[5];
  const float* aw   = (const float*)d_in[6];
  const float* ab   = (const float*)d_in[7];
  const float* tw   = (const float*)d_in[8];
  const float* tb   = (const float*)d_in[9];
  const float* ew   = (const float*)d_in[10];
  const float* eb   = (const float*)d_in[11];
  const float* m_w0 = (const float*)d_in[12];
  const float* m_w1 = (const float*)d_in[13];
  const float* m_ln = (const float*)d_in[14];
  float* out = (float*)d_out;

  char* p = (char*)d_ws;
  auto alloc = [&](size_t bytes) -> void* {
    void* q = (void*)p;
    p += (bytes + 255) & ~(size_t)255;
    return q;
  };
  const size_t DD = (size_t)DDIM * DDIM, ID = (size_t)IDIM * DDIM;

  u16_t* xbf  = (u16_t*)alloc((size_t)ROWS_ALL * DDIM * 2);
  u16_t* wqT  = (u16_t*)alloc(DD * 2);
  u16_t* wkT  = (u16_t*)alloc(DD * 2);
  u16_t* wvT  = (u16_t*)alloc(DD * 2);
  float* kf32 = (float*)alloc((size_t)ROWS_ALL * DDIM * 4);
  u16_t* kbf  = (u16_t*)alloc((size_t)ROWS_ALL * DDIM * 2);
  float* vf32 = (float*)alloc((size_t)ROWS_ALL * DDIM * 4);
  float* qf32 = (float*)alloc((size_t)ROWS_ALL * DDIM * 4);
  u16_t* qbf  = (u16_t*)alloc((size_t)ROWS_ALL * DDIM * 2);
  float* Mw0  = (float*)alloc(ID * 4);
  float* Sw0  = (float*)alloc(ID * 4);
  float* Mw1  = (float*)alloc(ID * 4);
  float* Sw1  = (float*)alloc(ID * 4);
  float* Mln  = (float*)alloc(DDIM * 4);
  float* Sln  = (float*)alloc(DDIM * 4);
  u16_t* w0T  = (u16_t*)alloc(ID * 2);   // [D,I]
  u16_t* w1bf = (u16_t*)alloc(ID * 2);   // [D,I]
  u16_t* w1T  = (u16_t*)alloc(ID * 2);   // [I,D]
  float* Araw = (float*)alloc((size_t)ROWS_ALL * IDIM * 4);
  u16_t* hbf  = (u16_t*)alloc((size_t)ROWS_ALL * IDIM * 2);
  float* yraw = (float*)alloc((size_t)ROWS_ALL * DDIM * 4); // also proj scratch
  u16_t* dybf = (u16_t*)alloc((size_t)ROWS_CH * DDIM * 2);
  u16_t* dyTa = (u16_t*)alloc((size_t)DDIM * ROWS_CH * 2);
  u16_t* dyTe = (u16_t*)alloc((size_t)DDIM * ROWS_CH * 2);
  float* dln  = (float*)alloc((size_t)ROWS_CH * DDIM * 4);
  float* dh   = (float*)alloc((size_t)ROWS_CH * IDIM * 4);
  u16_t* daTa = (u16_t*)alloc((size_t)IDIM * ROWS_CH * 2);
  u16_t* daTe = (u16_t*)alloc((size_t)IDIM * ROWS_CH * 2);
  float* U0a  = (float*)alloc(ID * 4);
  float* U0e  = (float*)alloc(ID * 4);
  float* U1a  = (float*)alloc(ID * 4);
  float* U1e  = (float*)alloc(ID * 4);
  float* ulna = (float*)alloc(DDIM * 4);
  float* ulne = (float*)alloc(DDIM * 4);
  float* gates= (float*)alloc((size_t)NCH * 256 * 4);

  // ---- one-time prep ----
  {
    size_t total = 3 * DD + 2 * ID + DDIM;
    prep_kernel<<<dim3((unsigned)((total + 255) / 256)), 256, 0, stream>>>(
        wq, wk, wv, m_w0, m_w1, m_ln, wqT, wkT, wvT,
        Mw0, Sw0, w0T, Mw1, Sw1, w1bf, w1T, Mln, Sln);
    pack_x_kernel<<<dim3((unsigned)(((size_t)ROWS_ALL * DDIM + 255) / 256)), 256, 0, stream>>>(x, xbf);
    gates_kernel<<<NCH, 256, 0, stream>>>(x, aw, ab, tw, tb, ew, eb, gates);
  }

  // ---- k / v / q projections (GEMM -> silu/RMS) ----
  gemm_bf16_nn<<<dim3(ROWS_ALL/128, DDIM/128), 256, 0, stream>>>(xbf, DDIM, wkT, DDIM, yraw, DDIM, DDIM);
  silu_rms_kernel<<<ROWS_ALL, 256, 0, stream>>>(yraw, knw, kf32, kbf, 1);
  gemm_bf16_nn<<<dim3(ROWS_ALL/128, DDIM/128), 256, 0, stream>>>(xbf, DDIM, wvT, DDIM, yraw, DDIM, DDIM);
  silu_rms_kernel<<<ROWS_ALL, 256, 0, stream>>>(yraw, nullptr, vf32, nullptr, 0);
  gemm_bf16_nn<<<dim3(ROWS_ALL/128, DDIM/128), 256, 0, stream>>>(xbf, DDIM, wqT, DDIM, yraw, DDIM, DDIM);
  silu_rms_kernel<<<ROWS_ALL, 256, 0, stream>>>(yraw, qnw, qf32, qbf, 1);

  // ---- sequential chunk scan ----
  for (int c = 0; c < NCH; ++c) {
    const float* gc = gates + (size_t)c * 256;
    const size_t row0 = (size_t)c * ROWS_CH;
    const u16_t* kbf_c = kbf + row0 * DDIM;
    // forward: A = k @ w0^T ; h = silu(A) ; y = h @ w1^T
    gemm_bf16_nn<<<dim3(ROWS_CH/128, IDIM/128), 256, 0, stream>>>(kbf_c, DDIM, w0T, IDIM, Araw, IDIM, DDIM);
    silu_elem_kernel<<<dim3((unsigned)(((size_t)ROWS_CH*IDIM + 255) / 256)), 256, 0, stream>>>(Araw, hbf, (size_t)ROWS_CH*IDIM);
    gemm_bf16_nn<<<dim3(ROWS_CH/128, DDIM/128), 256, 0, stream>>>(hbf, IDIM, w1T, DDIM, yraw, DDIM, IDIM);
    // backward through RMS head
    rowback_kernel<<<ROWS_CH, 256, 0, stream>>>(yraw, kf32 + row0 * DDIM, vf32 + row0 * DDIM,
                                                Mln, gc, dybf, dyTa, dyTe, dln);
    // dh = dy @ w1 ; da = dh * silu'(A)
    gemm_bf16_nn<<<dim3(ROWS_CH/128, IDIM/128), 256, 0, stream>>>(dybf, DDIM, w1bf, IDIM, dh, IDIM, DDIM);
    da_kernel<<<dim3((unsigned)(((size_t)ROWS_CH*IDIM + 255) / 256)), 256, 0, stream>>>(Araw, dh, gc, daTa, daTe);
    // weighted gradient GEMMs
    gemm_bf16_nn<<<dim3(DDIM/128, IDIM/128), 256, 0, stream>>>(dyTa, ROWS_CH, hbf, IDIM, U1a, IDIM, ROWS_CH);
    gemm_bf16_nn<<<dim3(DDIM/128, IDIM/128), 256, 0, stream>>>(dyTe, ROWS_CH, hbf, IDIM, U1e, IDIM, ROWS_CH);
    gemm_bf16_nn<<<dim3(IDIM/128, DDIM/128), 256, 0, stream>>>(daTa, ROWS_CH, kbf_c, DDIM, U0a, DDIM, ROWS_CH);
    gemm_bf16_nn<<<dim3(IDIM/128, DDIM/128), 256, 0, stream>>>(daTe, ROWS_CH, kbf_c, DDIM, U0e, DDIM, ROWS_CH);
    reduce_dln_kernel<<<dim3(2), 256, 0, stream>>>(dln, gc, ulna, ulne);
    // momentum / decay parameter update
    update_kernel<<<dim3(4098), 256, 0, stream>>>(gc, Mw0, Sw0, U0a, U0e, w0T,
                                                  Mw1, Sw1, U1a, U1e, w1bf, w1T,
                                                  Mln, Sln, ulna, ulne);
  }

  // ---- retrieval with final params ----
  gemm_bf16_nn<<<dim3(ROWS_ALL/128, IDIM/128), 256, 0, stream>>>(qbf, DDIM, w0T, IDIM, Araw, IDIM, DDIM);
  silu_elem_kernel<<<dim3((unsigned)(((size_t)ROWS_ALL*IDIM + 255) / 256)), 256, 0, stream>>>(Araw, hbf, (size_t)ROWS_ALL*IDIM);
  gemm_bf16_nn<<<dim3(ROWS_ALL/128, DDIM/128), 256, 0, stream>>>(hbf, IDIM, w1T, DDIM, yraw, DDIM, IDIM);
  final_kernel<<<ROWS_ALL, 256, 0, stream>>>(yraw, qf32, Mln, out);
}
